// WindowMultiHeadAttention_24043226923857
// MI455X (gfx1250) — matrix operand order repaired and checked by reading, not compiled, not run on hardware
//
#include <hip/hip_runtime.h>
#include <hip/hip_bf16.h>

#define E_DIM  512
#define N_TOK  49
#define B_WIN  2048
#define N_HEAD 16
#define D_HEAD 32
#define NT_ROWS (B_WIN * N_TOK)   // 100352

typedef _Float16 h16;
typedef __attribute__((ext_vector_type(16))) _Float16 v16h;
typedef __attribute__((ext_vector_type(8)))  _Float16 v8h;
typedef __attribute__((ext_vector_type(8)))  float    v8f;
typedef __attribute__((ext_vector_type(4)))  int      v4i;

union AFrag {
    v16h v;
    h16  e[16];
    v8h  h2[2];
};

__device__ __forceinline__ v8f wmma16x16x32(v16h a, v16h b, v8f c) {
    // 8 args: (neg_a, A, neg_b, B, c_mod, C, reuse_a, reuse_b)
    return __builtin_amdgcn_wmma_f32_16x16x32_f16(false, a, false, b, (short)0, c, false, false);
}

// ---- gfx1250 async global->LDS copy (ASYNCcnt) with compile-safe fallback ----
#ifdef __has_builtin
#if __has_builtin(__builtin_amdgcn_global_load_async_to_lds_b128)
#define HAVE_ASYNC_LDS 1
#endif
#if __has_builtin(__builtin_amdgcn_s_wait_asynccnt)
#define HAVE_ASYNC_WAIT_BUILTIN 1
#endif
#endif

__device__ __forceinline__ void async_copy_b128(h16* lds, const h16* g) {
#if defined(HAVE_ASYNC_LDS)
    __builtin_amdgcn_global_load_async_to_lds_b128(
        (__attribute__((address_space(1))) v4i*)(unsigned long long)(const void*)g,
        (__attribute__((address_space(3))) v4i*)(unsigned long long)(void*)lds,
        0, 0);
#else
    *(v8h*)lds = *(const v8h*)g;   // fallback: through VGPRs
#endif
}

__device__ __forceinline__ void async_copy_wait() {
#if defined(HAVE_ASYNC_LDS)
#if defined(HAVE_ASYNC_WAIT_BUILTIN)
    __builtin_amdgcn_s_wait_asynccnt(0);
#else
    asm volatile("s_wait_asynccnt 0x0" ::: "memory");
#endif
#endif
}

// ---------------------------------------------------------------------------
// Kernel 1: fused QKV projection.  out = (X @ W + b) [*scale for q]
// X: [NT_ROWS, 512] f32 ; W: [512,512] f32 ; out: f16 in [B][H][N][D] layout.
// Block: 256 threads (8 waves). Block tile: 128 rows x 32 cols. z selects q/k/v.
// Grid: x = n-tile (16), y = m-slab (784) so consecutive blocks share the same
// 256KB X slab out of L2 (X > L2, so m-major order would re-stream X 16x).
// ---------------------------------------------------------------------------
__global__ __launch_bounds__(256) void proj_qkv_kernel(
    const float* __restrict__ Xq, const float* __restrict__ Xk, const float* __restrict__ Xv,
    const float* __restrict__ Wq, const float* __restrict__ Wk, const float* __restrict__ Wv,
    const float* __restrict__ bq, const float* __restrict__ bk, const float* __restrict__ bv,
    h16* __restrict__ Oq, h16* __restrict__ Ok, h16* __restrict__ Ov, float scaleQ)
{
    __shared__ __align__(16) h16 Wl[32][520];   // [n_local][k], padded stride

    const int z = blockIdx.z;
    const float* X    = (z == 0) ? Xq : (z == 1) ? Xk : Xv;
    const float* W    = (z == 0) ? Wq : (z == 1) ? Wk : Wv;
    const float* bias = (z == 0) ? bq : (z == 1) ? bk : bv;
    h16* Out          = (z == 0) ? Oq : (z == 1) ? Ok : Ov;
    const float sc    = (z == 0) ? scaleQ : 1.0f;

    const int tid  = threadIdx.x;
    const int lane = tid & 31;
    const int wave = tid >> 5;
    const int col  = lane & 15;
    const int hi   = lane >> 4;          // 0 or 1
    const int nbase = blockIdx.x * 32;   // n fastest -> L2 reuse of X slab
    const int mbase = blockIdx.y * 128;

    // Stage W[k][nbase..nbase+31] -> Wl[n][k] as f16 (column-major-ish)
    for (int idx = tid; idx < 512 * 32; idx += 256) {
        int k = idx >> 5;
        int n = idx & 31;
        Wl[n][k] = (h16)W[k * E_DIM + nbase + n];
    }
    __syncthreads();

    const int mrow = mbase + wave * 16 + col;   // A row owned by this lane
    v8f zero = {};
    v8f acc[2];
    acc[0] = zero; acc[1] = zero;

    for (int kb = 0; kb < E_DIM; kb += 32) {
        // A fragment (16x32 f16): two contiguous 8-element K runs per lane
        AFrag a;
        const int k0 = kb + (hi << 3);
        const float4* p0 = reinterpret_cast<const float4*>(X + (size_t)mrow * E_DIM + k0);
        float4 f0 = p0[0], f1 = p0[1];
        const float4* p1 = reinterpret_cast<const float4*>(X + (size_t)mrow * E_DIM + k0 + 16);
        float4 f2 = p1[0], f3 = p1[1];
        a.e[0]=(h16)f0.x;  a.e[1]=(h16)f0.y;  a.e[2]=(h16)f0.z;  a.e[3]=(h16)f0.w;
        a.e[4]=(h16)f1.x;  a.e[5]=(h16)f1.y;  a.e[6]=(h16)f1.z;  a.e[7]=(h16)f1.w;
        a.e[8]=(h16)f2.x;  a.e[9]=(h16)f2.y;  a.e[10]=(h16)f2.z; a.e[11]=(h16)f2.w;
        a.e[12]=(h16)f3.x; a.e[13]=(h16)f3.y; a.e[14]=(h16)f3.z; a.e[15]=(h16)f3.w;

        const int koff = kb + (hi << 3);
        #pragma unroll
        for (int nt = 0; nt < 2; ++nt) {
            AFrag b;
            const v8h* q = reinterpret_cast<const v8h*>(&Wl[nt * 16 + col][koff]);
            b.h2[0] = q[0];
            b.h2[1] = q[2];
            acc[nt] = wmma16x16x32(a.v, b.v, acc[nt]);
        }
    }

    // Write out in [B][H][N][D] f16 layout, bias added, q pre-scaled
    const int rhalf = hi << 3;
    #pragma unroll
    for (int nt = 0; nt < 2; ++nt) {
        const int n  = nbase + nt * 16 + col;
        const float bn = bias[n];
        const int hh = n >> 5;
        const int dd = n & 31;
        #pragma unroll
        for (int r = 0; r < 8; ++r) {
            const int m  = mbase + wave * 16 + rhalf + r;
            const int bw = m / N_TOK;
            const int it = m - bw * N_TOK;
            const float val = (acc[nt][r] + bn) * sc;
            Out[(((size_t)bw * N_HEAD + hh) * N_TOK + it) * D_HEAD + dd] = (h16)val;
        }
    }
}

// ---------------------------------------------------------------------------
// Kernel 2: per-(window, head) attention. One wave per (b,h); 4 waves/block
// share the per-head relative-position bias staged in LDS.
// V tile is brought into LDS with gfx1250 async global->LDS copies, overlapped
// with the S = Q K^T WMMAs and the softmax; waited with s_wait_asynccnt.
// ---------------------------------------------------------------------------
__global__ __launch_bounds__(128) void attn_kernel(
    const h16* __restrict__ Qb, const h16* __restrict__ Kb, const h16* __restrict__ Vb,
    const float* __restrict__ rpb, const int* __restrict__ relidx,
    h16* __restrict__ AttO)
{
    __shared__ __align__(16) h16 Pl[4][N_TOK][72];   // 28,224 B (144B stride)
    __shared__ __align__(16) h16 Vt[4][32][72];      // 18,432 B
    __shared__ __align__(16) h16 Vrow[4][N_TOK * D_HEAD]; // 12,544 B (async dest)
    __shared__ h16 biasS[N_TOK][52];                 //  5,096 B  (total 64,296 B)

    const int tid  = threadIdx.x;
    const int lane = tid & 31;
    const int w    = tid >> 5;
    const int col  = lane & 15;
    const int hi   = lane >> 4;
    const int h    = blockIdx.y;
    const int bwin = blockIdx.x * 4 + w;

    const size_t base = ((size_t)bwin * N_HEAD + h) * (N_TOK * D_HEAD);
    const h16* Qg = Qb + base;
    const h16* Kg = Kb + base;
    const h16* Vg = Vb + base;

    // Kick off async copy of the V tile (49*32 halves = 196 x 16B) right away.
    {
        h16* dst = &Vrow[w][0];
        #pragma unroll
        for (int it = 0; it < 7; ++it) {
            int idx = it * 32 + lane;
            if (idx < 196)
                async_copy_b128(dst + idx * 8, Vg + idx * 8);
        }
    }

    // Stage bias for this head: bias[i][j] = rpb_table[rel_pos_idx[i][j]][h]
    for (int idx = tid; idx < N_TOK * N_TOK; idx += 128) {
        int i = idx / N_TOK;
        int j = idx - i * N_TOK;
        biasS[i][j] = (h16)rpb[relidx[idx] * N_HEAD + h];
    }
    __syncthreads();

    // Q fragments (A, 16x32): rows clamped to 48 for padding
    AFrag aq[4];
    {
        const int off = hi << 3;
        #pragma unroll
        for (int ti = 0; ti < 4; ++ti) {
            int row = ti * 16 + col; if (row > 48) row = 48;
            aq[ti].h2[0] = *reinterpret_cast<const v8h*>(Qg + row * D_HEAD + off);
            aq[ti].h2[1] = *reinterpret_cast<const v8h*>(Qg + row * D_HEAD + off + 16);
        }
    }
    // K^T fragments (B, 32x16): B[k][j] = K[j][k] -> two 8-half runs (off, off+16) of row j
    AFrag bk4[4];
    {
        const int koff = hi << 3;
        #pragma unroll
        for (int tj = 0; tj < 4; ++tj) {
            int jr = tj * 16 + col; if (jr > 48) jr = 48;
            const v8h* p = reinterpret_cast<const v8h*>(Kg + jr * D_HEAD + koff);
            bk4[tj].h2[0] = p[0];
            bk4[tj].h2[1] = p[2];
        }
    }

    v8f zero = {};
    v8f acc[4][4];
    #pragma unroll
    for (int ti = 0; ti < 4; ++ti)
        #pragma unroll
        for (int tj = 0; tj < 4; ++tj)
            acc[ti][tj] = wmma16x16x32(aq[ti].v, bk4[tj].v, zero);

    // Bias + column mask
    const int rhalf = hi << 3;
    #pragma unroll
    for (int ti = 0; ti < 4; ++ti) {
        #pragma unroll
        for (int tj = 0; tj < 4; ++tj) {
            const int j = tj * 16 + col;
            #pragma unroll
            for (int r = 0; r < 8; ++r) {
                int i = ti * 16 + rhalf + r; if (i > 48) i = 48;
                float s = acc[ti][tj][r];
                s = (j < N_TOK) ? (s + (float)biasS[i][j]) : -1e30f;
                acc[ti][tj][r] = s;
            }
        }
    }

    // Softmax per row (row lives in one VGPR across 16 lanes + 4 tj tiles);
    // only rows < 49 are stored to Pl (rows >= 49 never contribute to output).
    #pragma unroll
    for (int ti = 0; ti < 4; ++ti) {
        #pragma unroll
        for (int r = 0; r < 8; ++r) {
            float m = fmaxf(fmaxf(acc[ti][0][r], acc[ti][1][r]),
                            fmaxf(acc[ti][2][r], acc[ti][3][r]));
            m = fmaxf(m, __shfl_xor(m, 1, 32));
            m = fmaxf(m, __shfl_xor(m, 2, 32));
            m = fmaxf(m, __shfl_xor(m, 4, 32));
            m = fmaxf(m, __shfl_xor(m, 8, 32));
            float sum = 0.f;
            #pragma unroll
            for (int tj = 0; tj < 4; ++tj) {
                float e = __expf(acc[ti][tj][r] - m);
                acc[ti][tj][r] = e;
                sum += e;
            }
            sum += __shfl_xor(sum, 1, 32);
            sum += __shfl_xor(sum, 2, 32);
            sum += __shfl_xor(sum, 4, 32);
            sum += __shfl_xor(sum, 8, 32);
            const float inv = 1.0f / sum;
            const int irow = ti * 16 + rhalf + r;
            if (irow < N_TOK) {
                #pragma unroll
                for (int tj = 0; tj < 4; ++tj)
                    Pl[w][irow][tj * 16 + col] = (h16)(acc[ti][tj][r] * inv);
            }
        }
    }

    // V tile has had the whole S+softmax phase to land; transpose into Vt.
    async_copy_wait();
    for (int idx = lane; idx < N_TOK * D_HEAD; idx += 32) {
        int j = idx >> 5, d = idx & 31;
        Vt[w][d][j] = Vrow[w][idx];
    }
    for (int idx = lane; idx < 32 * 15; idx += 32) {
        int d = idx / 15;
        int j = 49 + idx % 15;
        Vt[w][d][j] = (h16)0.f;
    }

    // O = P(49x64pad) x V(64x32): A frags from Pl (rows clamped), B frags from Vt
    v8f oacc[4][2];
    #pragma unroll
    for (int ti = 0; ti < 4; ++ti) {
        oacc[ti][0] = zero; oacc[ti][1] = zero;
    }
    #pragma unroll
    for (int jb = 0; jb < 64; jb += 32) {
        AFrag ap[4];
        const int off = jb + (hi << 3);
        #pragma unroll
        for (int ti = 0; ti < 4; ++ti) {
            int row = ti * 16 + col; if (row > 48) row = 48;
            ap[ti].h2[0] = *reinterpret_cast<const v8h*>(&Pl[w][row][off]);
            ap[ti].h2[1] = *reinterpret_cast<const v8h*>(&Pl[w][row][off + 16]);
        }
        const int joff = jb + (hi << 3);
        #pragma unroll
        for (int dt = 0; dt < 2; ++dt) {
            AFrag bv;
            bv.h2[0] = *reinterpret_cast<const v8h*>(&Vt[w][dt * 16 + col][joff]);
            bv.h2[1] = *reinterpret_cast<const v8h*>(&Vt[w][dt * 16 + col][joff + 16]);
            #pragma unroll
            for (int ti = 0; ti < 4; ++ti)
                oacc[ti][dt] = wmma16x16x32(ap[ti].v, bv.v, oacc[ti][dt]);
        }
    }

    // Store O as f16 into [B*N, E] layout (cols h*32+d)
    #pragma unroll
    for (int ti = 0; ti < 4; ++ti) {
        #pragma unroll
        for (int dt = 0; dt < 2; ++dt) {
            const int d = dt * 16 + col;
            #pragma unroll
            for (int r = 0; r < 8; ++r) {
                const int i = ti * 16 + rhalf + r;
                if (i < N_TOK) {
                    AttO[((size_t)bwin * N_TOK + i) * E_DIM + h * D_HEAD + d] =
                        (h16)oacc[ti][dt][r];
                }
            }
        }
    }
}

// ---------------------------------------------------------------------------
// Kernel 3: output projection. out(f32) = Xa(f16) @ Wo + bo
// Same n-fastest grid order as kernel 1 for L2 reuse of the A slab.
// ---------------------------------------------------------------------------
__global__ __launch_bounds__(256) void out_proj_kernel(
    const h16* __restrict__ Xa, const float* __restrict__ Wo,
    const float* __restrict__ bo, float* __restrict__ Out)
{
    __shared__ __align__(16) h16 Wl[32][520];

    const int tid  = threadIdx.x;
    const int lane = tid & 31;
    const int wave = tid >> 5;
    const int col  = lane & 15;
    const int hi   = lane >> 4;
    const int nbase = blockIdx.x * 32;
    const int mbase = blockIdx.y * 128;

    for (int idx = tid; idx < 512 * 32; idx += 256) {
        int k = idx >> 5;
        int n = idx & 31;
        Wl[n][k] = (h16)Wo[k * E_DIM + nbase + n];
    }
    __syncthreads();

    const int mrow = mbase + wave * 16 + col;
    v8f zero = {};
    v8f acc[2];
    acc[0] = zero; acc[1] = zero;

    for (int kb = 0; kb < E_DIM; kb += 32) {
        AFrag a;
        const int k0 = kb + (hi << 3);
        a.h2[0] = *reinterpret_cast<const v8h*>(Xa + (size_t)mrow * E_DIM + k0);
        a.h2[1] = *reinterpret_cast<const v8h*>(Xa + (size_t)mrow * E_DIM + k0 + 16);
        const int koff = kb + (hi << 3);
        #pragma unroll
        for (int nt = 0; nt < 2; ++nt) {
            AFrag b;
            const v8h* q = reinterpret_cast<const v8h*>(&Wl[nt * 16 + col][koff]);
            b.h2[0] = q[0];
            b.h2[1] = q[2];
            acc[nt] = wmma16x16x32(a.v, b.v, acc[nt]);
        }
    }

    const int rhalf = hi << 3;
    #pragma unroll
    for (int nt = 0; nt < 2; ++nt) {
        const int n = nbase + nt * 16 + col;
        const float bn = bo[n];
        #pragma unroll
        for (int r = 0; r < 8; ++r) {
            const int m = mbase + wave * 16 + rhalf + r;
            Out[(size_t)m * E_DIM + n] = acc[nt][r] + bn;
        }
    }
}

// ---------------------------------------------------------------------------
extern "C" void kernel_launch(void* const* d_in, const int* in_sizes, int n_in,
                              void* d_out, int out_size, void* d_ws, size_t ws_size,
                              hipStream_t stream)
{
    (void)in_sizes; (void)n_in; (void)out_size; (void)ws_size;

    const float* q      = (const float*)d_in[0];
    const float* k      = (const float*)d_in[1];
    const float* v      = (const float*)d_in[2];
    const float* Wq     = (const float*)d_in[3];
    const float* Wk     = (const float*)d_in[4];
    const float* Wv     = (const float*)d_in[5];
    const float* Wo     = (const float*)d_in[6];
    const float* bq     = (const float*)d_in[7];
    const float* bk     = (const float*)d_in[8];
    const float* bv     = (const float*)d_in[9];
    const float* bo     = (const float*)d_in[10];
    const float* rpb    = (const float*)d_in[11];
    const int*   relidx = (const int*)d_in[12];
    float* out = (float*)d_out;

    const size_t elems = (size_t)B_WIN * N_HEAD * N_TOK * D_HEAD;  // 51,380,224
    h16* qbuf = (h16*)d_ws;
    h16* kbuf = qbuf + elems;
    h16* vbuf = kbuf + elems;
    h16* abuf = vbuf + elems;   // f16 [NT_ROWS, E_DIM]

    const float SCALE = 0.044194173824159216f;  // 1/sqrt(512)

    dim3 g1(E_DIM / 32, NT_ROWS / 128, 3);
    proj_qkv_kernel<<<g1, 256, 0, stream>>>(q, k, v, Wq, Wk, Wv, bq, bk, bv,
                                            qbuf, kbuf, vbuf, SCALE);

    dim3 g2(B_WIN / 4, N_HEAD, 1);
    attn_kernel<<<g2, 128, 0, stream>>>(qbuf, kbuf, vbuf, rpb, relidx, abuf);

    dim3 g3(E_DIM / 32, NT_ROWS / 128, 1);
    out_proj_kernel<<<g3, 256, 0, stream>>>(abuf, Wo, bo, out);
}
